// VRPGNNWithRouteEncoding_81853486727226
// MI455X (gfx1250) — compile-verified, matrix operand order checked by reading, not run on hardware
//
#include <hip/hip_runtime.h>
#include <hip/hip_bf16.h>

typedef __bf16 bf16_t;
typedef __attribute__((ext_vector_type(16))) __bf16 v16bf;
typedef __attribute__((ext_vector_type(8)))  float  v8f;

// ---------------------------------------------------------------- utilities

__device__ inline float wave_red_add(float v) {
#pragma unroll
  for (int off = 16; off > 0; off >>= 1) v += __shfl_xor(v, off, 32);
  return v;
}

// order-preserving float <-> uint key (for atomicMax-based segment max)
__device__ inline unsigned fkey(float f) {
  unsigned u = __float_as_uint(f);
  return (u & 0x80000000u) ? ~u : (u | 0x80000000u);
}
__device__ inline float funkey(unsigned k) {
  return (k & 0x80000000u) ? __uint_as_float(k & 0x7FFFFFFFu) : __uint_as_float(~k);
}

__global__ void fill_f32(float* __restrict__ p, long n, float v) {
  long i = (long)blockIdx.x * blockDim.x + threadIdx.x;
  if (i < n) p[i] = v;
}
__global__ void fill_u32(unsigned* __restrict__ p, long n, unsigned v) {
  long i = (long)blockIdx.x * blockDim.x + threadIdx.x;
  if (i < n) p[i] = v;
}

// ------------------------------------------------- WMMA GEMM: C = act(A@W+b)
// A: [Mrows,K] fp32 (or gathered [h[src]|h[dst]|ef] rows when GATHER=1)
// W: [K,Ncols] fp32 row-major. Internally bf16, f32 accumulation.
// Block = 128 threads (4 waves); tile 64x64; wave w -> rows 16w..16w+15.
// Double-buffered LDS; per-thread float4 staging; templated epilogue.
// Requires: K, Ncols multiples of 16; Ncols multiple of 64 per grid sizing.

#define GEMM_TM 64
#define GEMM_TN 64
#define GEMM_TK 32

template <int GATHER, int ACT, int HASBIAS>
__global__ __launch_bounds__(128)
void gemm_wmma_kernel(const float* __restrict__ A, const float* __restrict__ W,
                      const float* __restrict__ bias, float* __restrict__ C,
                      int Mrows, int K, int Ncols,
                      const int* __restrict__ gsrc, const int* __restrict__ gdst,
                      const float* __restrict__ hnode, const float* __restrict__ efp) {
  __shared__ bf16_t As[2][GEMM_TM][GEMM_TK + 2];
  __shared__ bf16_t Bt[2][GEMM_TN][GEMM_TK + 2];

  const int tid  = threadIdx.x;
  const int lane = tid & 31;
  const int wave = tid >> 5;       // 0..3
  const int half = lane >> 4;      // 0..1
  const int lrow = lane & 15;

  const int row0 = blockIdx.x * GEMM_TM;
  const int col0 = blockIdx.y * GEMM_TN;

  // staging assignments (fixed per thread)
  const int ar  = tid >> 1;              // A row 0..63
  const int ac0 = (tid & 1) << 4;        // A col chunk {0,16}
  const int bk  = tid & 31;              // B k-row 0..31
  const int bn0 = (tid >> 5) << 4;       // B n chunk {0,16,32,48}

  const int agrow = row0 + ar;
  const bool arow_in = (agrow < Mrows);

  // resolve gather indirection once per thread (chunks never straddle segments)
  const float* asrc_base = nullptr;  // base of this thread's A source row (excl. k)
  if (arow_in) {
    if (!GATHER) {
      asrc_base = A + (long)agrow * K;
    }
  }
  int g_s = 0, g_d = 0;
  if (GATHER && arow_in) { g_s = gsrc[agrow]; g_d = gdst[agrow]; }

  auto stage = [&](int buf, int k0) {
    // ---- A tile: one thread loads 16 contiguous floats of one row ----
    {
      const int gk0 = k0 + ac0;
      float4 v[4];
      bool kin = (gk0 + 16 <= K) && arow_in;
      if (kin) {
        const float* p;
        if (GATHER) {
          if (gk0 < 128)      p = hnode + (long)g_s * 128 + gk0;
          else if (gk0 < 256) p = hnode + (long)g_d * 128 + (gk0 - 128);
          else                p = efp + (long)agrow * 128 + (gk0 - 256);
        } else {
          p = asrc_base + gk0;
        }
        const float4* p4 = (const float4*)p;
#pragma unroll
        for (int t = 0; t < 4; ++t) v[t] = p4[t];
      } else {
#pragma unroll
        for (int t = 0; t < 4; ++t) v[t] = make_float4(0.f, 0.f, 0.f, 0.f);
      }
#pragma unroll
      for (int t = 0; t < 4; ++t) {
        As[buf][ar][ac0 + 4 * t + 0] = (bf16_t)v[t].x;
        As[buf][ar][ac0 + 4 * t + 1] = (bf16_t)v[t].y;
        As[buf][ar][ac0 + 4 * t + 2] = (bf16_t)v[t].z;
        As[buf][ar][ac0 + 4 * t + 3] = (bf16_t)v[t].w;
      }
    }
    // ---- B tile: one thread loads 16 contiguous floats of one K-row ----
    {
      const int gk = k0 + bk;
      float4 v[4];
      if (gk < K) {
        const float4* p4 = (const float4*)(W + (long)gk * Ncols + col0 + bn0);
#pragma unroll
        for (int t = 0; t < 4; ++t) v[t] = p4[t];
      } else {
#pragma unroll
        for (int t = 0; t < 4; ++t) v[t] = make_float4(0.f, 0.f, 0.f, 0.f);
      }
#pragma unroll
      for (int t = 0; t < 4; ++t) {
        Bt[buf][bn0 + 4 * t + 0][bk] = (bf16_t)v[t].x;
        Bt[buf][bn0 + 4 * t + 1][bk] = (bf16_t)v[t].y;
        Bt[buf][bn0 + 4 * t + 2][bk] = (bf16_t)v[t].z;
        Bt[buf][bn0 + 4 * t + 3][bk] = (bf16_t)v[t].w;
      }
    }
  };

  v8f acc[4] = {};
  const int nk = (K + GEMM_TK - 1) / GEMM_TK;

  stage(0, 0);
  __syncthreads();

  for (int i = 0; i < nk; ++i) {
    const int buf = i & 1;
    if (i + 1 < nk) stage(buf ^ 1, (i + 1) * GEMM_TK);
    // prefetch tile i+2 (L2 warm-up; global_prefetch_b8)
    if (i + 2 < nk) {
      const int pk = (i + 2) * GEMM_TK;
      if (!GATHER && arow_in) __builtin_prefetch(asrc_base + pk + ac0, 0, 1);
      __builtin_prefetch(W + (long)(pk + bk) * Ncols + col0 + bn0, 0, 1);
    }

    // A fragment: ISA 16-bit 16x32 layout
    v16bf af;
#pragma unroll
    for (int r = 0; r < 8; ++r) {
      int kb = ((r >> 2) << 4) + (half << 3) + ((r & 3) << 1);
      af[2 * r]     = As[buf][wave * 16 + lrow][kb];
      af[2 * r + 1] = As[buf][wave * 16 + lrow][kb + 1];
    }
#pragma unroll
    for (int j = 0; j < 4; ++j) {
      v16bf bfv;
#pragma unroll
      for (int r = 0; r < 8; ++r) {
        int kb = ((r >> 2) << 4) + (half << 3) + ((r & 3) << 1);
        bfv[2 * r]     = Bt[buf][j * 16 + lrow][kb];
        bfv[2 * r + 1] = Bt[buf][j * 16 + lrow][kb + 1];
      }
      acc[j] = __builtin_amdgcn_wmma_f32_16x16x32_bf16(false, af, false, bfv,
                                                       (short)0, acc[j], false, false);
    }
    __syncthreads();
  }

  // epilogue: C/D layout -> lane {0-15: M=v, 16-31: M=v+8}, N = lane&15
#pragma unroll
  for (int j = 0; j < 4; ++j) {
    int gcol = col0 + j * 16 + lrow;
    float bv = HASBIAS ? bias[gcol] : 0.f;
#pragma unroll
    for (int v = 0; v < 8; ++v) {
      int m    = v + (half ? 8 : 0);
      int grow = row0 + wave * 16 + m;
      if (grow < Mrows) {
        float x = acc[j][v] + bv;
        if (ACT == 1) x = fmaxf(x, 0.f);
        C[(long)grow * Ncols + gcol] = x;
      }
    }
  }
}

// ---------------------------------------------- GAT attention helper kernels

__global__ __launch_bounds__(128)
void attn_coeff_kernel(const float* __restrict__ h, const float* __restrict__ a_src,
                       const float* __restrict__ a_dst, float* __restrict__ es,
                       float* __restrict__ ed, int N, int Hh) {
  int node = blockIdx.x * 4 + (threadIdx.x >> 5);
  int lane = threadIdx.x & 31;
  if (node >= N) return;
  const float* row = h + (long)node * Hh * 128;
  for (int hh = 0; hh < Hh; ++hh) {
    float ss = 0.f, sd = 0.f;
#pragma unroll
    for (int t = 0; t < 4; ++t) {
      int d = lane + t * 32;
      float x = row[hh * 128 + d];
      ss += x * a_src[hh * 128 + d];
      sd += x * a_dst[hh * 128 + d];
    }
    ss = wave_red_add(ss);
    sd = wave_red_add(sd);
    if (lane == 0) { es[node * Hh + hh] = ss; ed[node * Hh + hh] = sd; }
  }
}

__global__ void att_max_kernel(const float* __restrict__ es, const float* __restrict__ ed,
                               const int* __restrict__ src, const int* __restrict__ dst,
                               float* __restrict__ ebuf, unsigned* __restrict__ mkeys,
                               int E, int N, int Hh) {
  int i = blockIdx.x * blockDim.x + threadIdx.x;
  if (i >= E + N) return;
  int s = (i < E) ? src[i] : (i - E);
  int d = (i < E) ? dst[i] : (i - E);
  for (int hh = 0; hh < Hh; ++hh) {
    float e = es[s * Hh + hh] + ed[d * Hh + hh];
    e = (e > 0.f) ? e : 0.2f * e;
    ebuf[(long)i * Hh + hh] = e;
    atomicMax(&mkeys[d * Hh + hh], fkey(e));
  }
}

__global__ void att_expsum_kernel(float* __restrict__ ebuf, const unsigned* __restrict__ mkeys,
                                  float* __restrict__ denom, const int* __restrict__ dst,
                                  int E, int N, int Hh) {
  int i = blockIdx.x * blockDim.x + threadIdx.x;
  if (i >= E + N) return;
  int d = (i < E) ? dst[i] : (i - E);
  for (int hh = 0; hh < Hh; ++hh) {
    float m = funkey(mkeys[d * Hh + hh]);
    float p = __expf(ebuf[(long)i * Hh + hh] - m);
    ebuf[(long)i * Hh + hh] = p;
    atomicAdd(&denom[d * Hh + hh], p);
  }
}

__global__ __launch_bounds__(128)
void aggregate_kernel(const float* __restrict__ p, const float* __restrict__ denom,
                      const float* __restrict__ hlin, const int* __restrict__ src,
                      const int* __restrict__ dst, float* __restrict__ out,
                      int E, int N, int Hh) {
  int wid  = blockIdx.x * 4 + (threadIdx.x >> 5);
  int lane = threadIdx.x & 31;
  if (wid >= E + N) return;
  int s = (wid < E) ? src[wid] : (wid - E);
  int d = (wid < E) ? dst[wid] : (wid - E);
  int dim = Hh * 128;
  for (int hh = 0; hh < Hh; ++hh) {
    float alpha = p[(long)wid * Hh + hh] / (denom[d * Hh + hh] + 1e-16f);
#pragma unroll
    for (int t = 0; t < 4; ++t) {
      int f = hh * 128 + lane + t * 32;
      atomicAdd(&out[(long)d * dim + f], hlin[(long)s * dim + f] * alpha);
    }
  }
}

// x = act(x + bias)   act: 1=relu, 2=elu
__global__ void bias_act_kernel(float* __restrict__ x, const float* __restrict__ bias,
                                long n, int dim, int act) {
  long i = (long)blockIdx.x * blockDim.x + threadIdx.x;
  if (i >= n) return;
  float v = x[i] + bias[i % dim];
  if (act == 1) v = fmaxf(v, 0.f);
  else if (act == 2) v = (v > 0.f) ? v : (__expf(v) - 1.f);
  x[i] = v;
}

// ----------------------------- route predictor + route-aware embedding update
__global__ __launch_bounds__(128)
void route_kernel(float* __restrict__ h, const float* __restrict__ Wr1,
                  const float* __restrict__ br1, const float* __restrict__ Wr2,
                  const float* __restrict__ br2, const float* __restrict__ route_emb,
                  float* __restrict__ rl_out, int N) {
  __shared__ float hrow[4][128];
  int wave = threadIdx.x >> 5, lane = threadIdx.x & 31;
  int n = blockIdx.x * 4 + wave;
  bool valid = (n < N);
  float x[4];
  if (valid) {
#pragma unroll
    for (int t = 0; t < 4; ++t) {
      x[t] = h[(long)n * 128 + lane + t * 32];
      hrow[wave][lane + t * 32] = x[t];
    }
  }
  __syncthreads();
  if (!valid) return;

  float ha = br1[lane], hb = br1[lane + 32];
  for (int k = 0; k < 128; ++k) {
    float hv = hrow[wave][k];
    ha += hv * Wr1[k * 64 + lane];
    hb += hv * Wr1[k * 64 + lane + 32];
  }
  ha = fmaxf(ha, 0.f);
  hb = fmaxf(hb, 0.f);

  float lg[5];
#pragma unroll
  for (int j = 0; j < 5; ++j) {
    float pj = ha * Wr2[lane * 5 + j] + hb * Wr2[(lane + 32) * 5 + j];
    lg[j] = wave_red_add(pj) + br2[j];
  }
  if (lane < 5) rl_out[(long)n * 5 + lane] = lg[lane];

  float m = lg[0];
#pragma unroll
  for (int j = 1; j < 5; ++j) m = fmaxf(m, lg[j]);
  float pr[5], sum = 0.f;
#pragma unroll
  for (int j = 0; j < 5; ++j) { pr[j] = __expf(lg[j] - m); sum += pr[j]; }
  float inv = 1.f / sum;
#pragma unroll
  for (int t = 0; t < 4; ++t) {
    int f = lane + t * 32;
    float dlt = 0.f;
#pragma unroll
    for (int j = 0; j < 5; ++j) dlt += pr[j] * inv * route_emb[j * 128 + f];
    h[(long)n * 128 + f] = x[t] + dlt;
  }
}

// -------------------------------- fused edge encoder: relu(ea@We1+be1)@We2+be2
#define EE_TILE 64
__global__ __launch_bounds__(128)
void edge_encoder_kernel(const float* __restrict__ edge_attr, const float* __restrict__ We1,
                         const float* __restrict__ be1, const float* __restrict__ We2,
                         const float* __restrict__ be2, float* __restrict__ ef, int E) {
  __shared__ float  We1s[8][128];
  __shared__ float  ea[EE_TILE][8];
  __shared__ bf16_t hid[EE_TILE][132];
  __shared__ bf16_t Wt2[128][132];

  const int tid  = threadIdx.x;
  const int lane = tid & 31;
  const int wave = tid >> 5;
  const int half = lane >> 4;
  const int lrow = lane & 15;
  const int e0 = blockIdx.x * EE_TILE;

  for (int idx = tid; idx < 8 * 128; idx += 128) We1s[idx >> 7][idx & 127] = We1[idx];
  // We2 staged transposed with float4 reads: thread owns k-row = tid
  {
    const float4* p4 = (const float4*)(We2 + (long)tid * 128);
#pragma unroll
    for (int t = 0; t < 32; ++t) {
      float4 v = p4[t];
      Wt2[4 * t + 0][tid] = (bf16_t)v.x;
      Wt2[4 * t + 1][tid] = (bf16_t)v.y;
      Wt2[4 * t + 2][tid] = (bf16_t)v.z;
      Wt2[4 * t + 3][tid] = (bf16_t)v.w;
    }
  }
  for (int idx = tid; idx < EE_TILE * 8; idx += 128) {
    int e = idx >> 3, k = idx & 7;
    int ge = e0 + e;
    ea[e][k] = (ge < E) ? edge_attr[(long)ge * 8 + k] : 0.f;
  }
  __syncthreads();

  // hidden = relu(ea @ We1 + be1): each thread owns output column j = tid
  float b1v = be1[tid];
  for (int e = 0; e < EE_TILE; ++e) {
    float s = b1v;
#pragma unroll
    for (int k = 0; k < 8; ++k) s += ea[e][k] * We1s[k][tid];
    hid[e][tid] = (bf16_t)fmaxf(s, 0.f);
  }
  __syncthreads();

  // hidden[64,128] @ We2[128,128] via WMMA; wave -> 16 rows x 128 cols
  v8f acc[8] = {};
  for (int kk0 = 0; kk0 < 128; kk0 += 32) {
    v16bf af;
#pragma unroll
    for (int r = 0; r < 8; ++r) {
      int kb = kk0 + ((r >> 2) << 4) + (half << 3) + ((r & 3) << 1);
      af[2 * r]     = hid[wave * 16 + lrow][kb];
      af[2 * r + 1] = hid[wave * 16 + lrow][kb + 1];
    }
#pragma unroll
    for (int j = 0; j < 8; ++j) {
      v16bf bfv;
#pragma unroll
      for (int r = 0; r < 8; ++r) {
        int kb = kk0 + ((r >> 2) << 4) + (half << 3) + ((r & 3) << 1);
        bfv[2 * r]     = Wt2[j * 16 + lrow][kb];
        bfv[2 * r + 1] = Wt2[j * 16 + lrow][kb + 1];
      }
      acc[j] = __builtin_amdgcn_wmma_f32_16x16x32_bf16(false, af, false, bfv,
                                                       (short)0, acc[j], false, false);
    }
  }
#pragma unroll
  for (int j = 0; j < 8; ++j) {
#pragma unroll
    for (int v = 0; v < 8; ++v) {
      int m  = v + (half ? 8 : 0);
      int ge = e0 + wave * 16 + m;
      int gc = j * 16 + lrow;
      if (ge < E) ef[(long)ge * 128 + gc] = acc[j][v] + be2[gc];
    }
  }
}

// ------------------------------- final 64->2 classifier head + log_softmax
__global__ __launch_bounds__(128)
void edge_logits_kernel(const float* __restrict__ z2, const float* __restrict__ Wc3,
                        const float* __restrict__ bc3, float* __restrict__ out, int E) {
  int e    = blockIdx.x * 4 + (threadIdx.x >> 5);
  int lane = threadIdx.x & 31;
  if (e >= E) return;
  float a = z2[(long)e * 64 + lane];
  float b = z2[(long)e * 64 + lane + 32];
  float l0 = wave_red_add(a * Wc3[lane * 2 + 0] + b * Wc3[(lane + 32) * 2 + 0]) + bc3[0];
  float l1 = wave_red_add(a * Wc3[lane * 2 + 1] + b * Wc3[(lane + 32) * 2 + 1]) + bc3[1];
  if (lane == 0) {
    float m   = fmaxf(l0, l1);
    float lse = m + __logf(__expf(l0 - m) + __expf(l1 - m));
    out[(long)e * 2 + 0] = l0 - lse;
    out[(long)e * 2 + 1] = l1 - lse;
  }
}

// ---------------------------------------------------------------- launcher

extern "C" void kernel_launch(void* const* d_in, const int* in_sizes, int n_in,
                              void* d_out, int out_size, void* d_ws, size_t ws_size,
                              hipStream_t stream) {
  (void)n_in; (void)out_size; (void)ws_size;
  const float* x     = (const float*)d_in[0];
  const int*   srcp  = (const int*)d_in[1];
  const int*   dstp  = (const int*)d_in[2];
  const float* eattr = (const float*)d_in[3];
  const float* W1  = (const float*)d_in[4];
  const float* as1 = (const float*)d_in[5];
  const float* ad1 = (const float*)d_in[6];
  const float* b1  = (const float*)d_in[7];
  const float* W2  = (const float*)d_in[8];
  const float* as2 = (const float*)d_in[9];
  const float* ad2 = (const float*)d_in[10];
  const float* b2  = (const float*)d_in[11];
  const float* W3  = (const float*)d_in[12];
  const float* as3 = (const float*)d_in[13];
  const float* ad3 = (const float*)d_in[14];
  const float* b3  = (const float*)d_in[15];
  const float* Wf  = (const float*)d_in[16];
  const float* asf = (const float*)d_in[17];
  const float* adf = (const float*)d_in[18];
  const float* bf  = (const float*)d_in[19];
  const float* route_emb = (const float*)d_in[20];
  const float* We1 = (const float*)d_in[21];
  const float* be1 = (const float*)d_in[22];
  const float* We2 = (const float*)d_in[23];
  const float* be2 = (const float*)d_in[24];
  const float* Wr1 = (const float*)d_in[25];
  const float* br1 = (const float*)d_in[26];
  const float* Wr2 = (const float*)d_in[27];
  const float* br2 = (const float*)d_in[28];
  const float* Wc1 = (const float*)d_in[29];
  const float* bc1 = (const float*)d_in[30];
  const float* Wc2 = (const float*)d_in[31];
  const float* bc2 = (const float*)d_in[32];
  const float* Wc3 = (const float*)d_in[33];
  const float* bc3 = (const float*)d_in[34];

  const int Nn = in_sizes[0] / 16;   // 30000
  const int Ee = in_sizes[1];        // 240000

  // workspace layout (fp32 elements)
  float* w = (float*)d_ws;
  long off = 0;
  float*    hA    = w + off; off += (long)Nn * 512;
  float*    hB    = w + off; off += (long)Nn * 512;
  float*    es    = w + off; off += (long)Nn * 4;
  float*    ed    = w + off; off += (long)Nn * 4;
  float*    ebuf  = w + off; off += (long)(Ee + Nn) * 4;
  unsigned* mkeys = (unsigned*)(w + off); off += (long)Nn * 4;
  float*    denom = w + off; off += (long)Nn * 4;
  float*    ef    = w + off; off += (long)Ee * 128;
  float*    z1    = w + off; off += (long)Ee * 128;
  float*    z2    = w + off; off += (long)Ee * 64;

  auto gdim = [](int Mr, int Nc) {
    return dim3((Mr + GEMM_TM - 1) / GEMM_TM, (Nc + GEMM_TN - 1) / GEMM_TN);
  };

  auto run_att = [&](const float* hlin, const float* asrc, const float* adst,
                     const float* bias, float* outbuf, int Hh, int actcode) {
    int dim = Hh * 128;
    long nh = (long)Nn * Hh;
    attn_coeff_kernel<<<(Nn + 3) / 4, 128, 0, stream>>>(hlin, asrc, adst, es, ed, Nn, Hh);
    fill_u32<<<(unsigned)((nh + 255) / 256), 256, 0, stream>>>(mkeys, nh, 0x007FFFFFu); // key(-inf)
    fill_f32<<<(unsigned)((nh + 255) / 256), 256, 0, stream>>>(denom, nh, 0.f);
    int tot = Ee + Nn;
    att_max_kernel<<<(tot + 255) / 256, 256, 0, stream>>>(es, ed, srcp, dstp, ebuf, mkeys, Ee, Nn, Hh);
    att_expsum_kernel<<<(tot + 255) / 256, 256, 0, stream>>>(ebuf, mkeys, denom, dstp, Ee, Nn, Hh);
    long nd = (long)Nn * dim;
    fill_f32<<<(unsigned)((nd + 255) / 256), 256, 0, stream>>>(outbuf, nd, 0.f);
    aggregate_kernel<<<(tot + 3) / 4, 128, 0, stream>>>(ebuf, denom, hlin, srcp, dstp, outbuf, Ee, Nn, Hh);
    bias_act_kernel<<<(unsigned)((nd + 255) / 256), 256, 0, stream>>>(outbuf, bias, nd, dim, actcode);
  };

  // ---- GAT stack (plain GEMMs: no bias, no act in GEMM) ----
  gemm_wmma_kernel<0, 0, 0><<<gdim(Nn, 512), 128, 0, stream>>>(
      x, W1, nullptr, hA, Nn, 16, 512, nullptr, nullptr, nullptr, nullptr);
  run_att(hA, as1, ad1, b1, hB, 4, 2);
  gemm_wmma_kernel<0, 0, 0><<<gdim(Nn, 512), 128, 0, stream>>>(
      hB, W2, nullptr, hA, Nn, 512, 512, nullptr, nullptr, nullptr, nullptr);
  run_att(hA, as2, ad2, b2, hB, 4, 2);
  gemm_wmma_kernel<0, 0, 0><<<gdim(Nn, 512), 128, 0, stream>>>(
      hB, W3, nullptr, hA, Nn, 512, 512, nullptr, nullptr, nullptr, nullptr);
  run_att(hA, as3, ad3, b3, hB, 4, 2);
  gemm_wmma_kernel<0, 0, 0><<<gdim(Nn, 128), 128, 0, stream>>>(
      hB, Wf, nullptr, hA, Nn, 512, 128, nullptr, nullptr, nullptr, nullptr);
  run_att(hA, asf, adf, bf, hB, 1, 1);   // hB = final node emb [N,128]

  float* out = (float*)d_out;
  // route head: writes route_logits to tail of d_out, updates hB in place
  route_kernel<<<(Nn + 3) / 4, 128, 0, stream>>>(hB, Wr1, br1, Wr2, br2, route_emb,
                                                 out + (long)Ee * 2, Nn);
  // edge feature encoder
  edge_encoder_kernel<<<(Ee + EE_TILE - 1) / EE_TILE, 128, 0, stream>>>(
      eattr, We1, be1, We2, be2, ef, Ee);

  // edge classifier: gathered [h[src]|h[dst]|ef] @ Wc1 -> relu -> @Wc2 -> relu -> head
  gemm_wmma_kernel<1, 1, 1><<<gdim(Ee, 128), 128, 0, stream>>>(
      nullptr, Wc1, bc1, z1, Ee, 384, 128, srcp, dstp, hB, ef);
  gemm_wmma_kernel<0, 1, 1><<<gdim(Ee, 64), 128, 0, stream>>>(
      z1, Wc2, bc2, z2, Ee, 128, 64, nullptr, nullptr, nullptr, nullptr);
  edge_logits_kernel<<<(Ee + 3) / 4, 128, 0, stream>>>(z2, Wc3, bc3, out, Ee);
}
